// MultiheadAttention_21380347200198
// MI455X (gfx1250) — matrix operand order repaired and checked by reading, not compiled, not run on hardware
//
#include <hip/hip_runtime.h>
#include <hip/hip_bf16.h>

typedef _Float16 half_t;
typedef __attribute__((ext_vector_type(16))) _Float16 v16h;
typedef __attribute__((ext_vector_type(8)))  _Float16 v8h;
typedef __attribute__((ext_vector_type(8)))  float    v8f;
typedef __attribute__((ext_vector_type(4)))  unsigned int u32x4;
typedef __attribute__((ext_vector_type(8)))  int          i32x8;
typedef __attribute__((ext_vector_type(4)))  int          i32x4;

#define DEVINLINE __device__ __forceinline__

#if __has_builtin(__builtin_amdgcn_tensor_load_to_lds) && __has_builtin(__builtin_amdgcn_s_wait_tensorcnt)
#define HAVE_TDM 1
#else
#define HAVE_TDM 0
#endif

constexpr int B_ = 8, S_ = 1024, E_ = 1024, H_ = 16, D_ = 64;
constexpr float NEG_INF_ = -3.402823466e38f;

DEVINLINE v16h cat16(v8h lo, v8h hi) {
  union { v16h v; v8h h[2]; } u;
  u.h[0] = lo; u.h[1] = hi;
  return u.v;
}

// A-fragment (16x32 f16, MxK) per ISA 7.12.2:
// lane&15 = M row; halves 0..7 -> K = 8*(lane>>4)+0..7 ; halves 8..15 -> +16.
DEVINLINE v16h load_a_frag(const half_t* base, int ld) {
  const int lane = threadIdx.x & 31;
  const half_t* row = base + (size_t)(lane & 15) * ld + ((lane >> 4) << 3);
  v8h lo = *(const v8h*)(row);
  v8h hi = *(const v8h*)(row + 16);
  return cat16(lo, hi);
}

// B-fragment (32x16 f16, KxN) gathered from an n-major tile T[n][k] (ldn in halves):
// lane&15 = N col; halves 0..7 -> K = 8*(lane>>4)+0..7 ; halves 8..15 -> +16.
DEVINLINE v16h load_b_frag(const half_t* base, int ldn) {
  const int lane = threadIdx.x & 31;
  const half_t* col = base + (size_t)(lane & 15) * ldn + ((lane >> 4) << 3);
  v8h lo = *(const v8h*)(col);
  v8h hi = *(const v8h*)(col + 16);
  return cat16(lo, hi);
}

DEVINLINE v8f wmma_f16(v16h a, v16h b, v8f c) {
  return __builtin_amdgcn_wmma_f32_16x16x32_f16(
      /*neg_a=*/false, a, /*neg_b=*/false, b,
      /*c_mod=*/(short)0, c, /*reuse_a=*/false, /*reuse_b=*/false);
}

#if HAVE_TDM
// TDM: DMA a 2D f16 tile (tile_d0 x tile_d1, row stride stride_d0 elems) from
// global into LDS, per CDNA5 ISA ch.8 descriptor layout (D# groups 0/1).
// This toolchain exposes the 6-arg builtin (clang-23 / therock-10.0 form).
DEVINLINE void tdm_load_2d_f16(unsigned int lds_off, const void* gaddr,
                               unsigned int tile_d0, unsigned int tile_d1,
                               unsigned int stride_d0) {
  unsigned long long ga = (unsigned long long)(size_t)gaddr;
  u32x4 g0;
  g0[0] = 1u;                                   // count=1, user mode, no gather
  g0[1] = lds_off;                              // lds_addr (bytes)
  g0[2] = (unsigned int)(ga & 0xFFFFFFFFu);     // global_addr[31:0]
  g0[3] = (unsigned int)((ga >> 32) & 0x1FFFFFFu) | (2u << 30);  // addr[56:32] | type=2
  i32x8 g1;
  g1[0] = (int)(1u << 16);                      // wg_mask=0, data_size=1 (2 bytes)
  g1[1] = (int)((tile_d0 & 0xFFFFu) << 16);     // tensor_dim0[15:0]  (bits 63:48)
  g1[2] = (int)(((tile_d0 >> 16) & 0xFFFFu) |   // tensor_dim0[31:16] (bits 79:64)
                ((tile_d1 & 0xFFFFu) << 16));   // tensor_dim1[15:0]  (bits 95:80)
  g1[3] = (int)(((tile_d1 >> 16) & 0xFFFFu) |   // tensor_dim1[31:16]
                ((tile_d0 & 0xFFFFu) << 16));   // tile_dim0 (bits 127:112)
  g1[4] = (int)(tile_d1 & 0xFFFFu);             // tile_dim1 (143:128), tile_dim2=0
  g1[5] = (int)stride_d0;                       // tensor_dim0_stride[31:0]
  g1[6] = 0;                                    // stride hi + dim1_stride lo
  g1[7] = 0;
  i32x4 z4 = {0, 0, 0, 0};                      // groups 2/3 unused (2D tensor)
  i32x8 z8 = {0, 0, 0, 0, 0, 0, 0, 0};
  __builtin_amdgcn_tensor_load_to_lds(g0, g1, z4, z4, z8, 0);
}
#endif

// ---------------------------------------------------------------------------
// C[M,N] = (A[M,K] @ W[N,K]^T + bias[N]) * scale
// Block: 256 thr (8 waves), tile 128x64, BK=32. Waves 4(M)x2(N), each wave
// computes 32x32 of C -> 4 WMMA per k-step (2 A-frags x 2 B-frags).
// ---------------------------------------------------------------------------
template <bool A_HALF, bool OUT_HALF>
__global__ __launch_bounds__(256)
void gemm_xwt(const void* __restrict__ Ap, const float* __restrict__ W,
              const float* __restrict__ bias, float scale,
              void* __restrict__ Outp, int M, int N, int K) {
  __shared__ __align__(16) half_t As[128][32];
  __shared__ __align__(16) half_t Bs[64][32];   // Bs[n][k] = W[n0+n][k0+k]

  const int m0 = blockIdx.x * 128;
  const int n0 = blockIdx.y * 64;
  const int t = threadIdx.x;
  const int lane = t & 31;
  const int wave = t >> 5;
  const int wm = wave & 3;   // 0..3 -> 32-row subtile
  const int wn = wave >> 2;  // 0..1 -> 32-col subtile

  const int arow = t >> 1;         // 0..127
  const int akq  = (t & 1) << 4;   // 0,16 (16 halves per thread)
  const int brow = t >> 2;         // 0..63
  const int bkq  = (t & 3) << 3;   // 0,8,16,24 (8 halves per thread)

  v8f c00 = {}, c01 = {}, c10 = {}, c11 = {};

  for (int k0 = 0; k0 < K; k0 += 32) {
    // ---- stage A tile (fp32->f16 convert on the fly if needed) ----
    if constexpr (A_HALF) {
      const half_t* Ah = (const half_t*)Ap + (size_t)(m0 + arow) * K + k0 + akq;
      *(v8h*)&As[arow][akq]     = *(const v8h*)(Ah);
      *(v8h*)&As[arow][akq + 8] = *(const v8h*)(Ah + 8);
    } else {
      const float* Af = (const float*)Ap + (size_t)(m0 + arow) * K + k0 + akq;
      #pragma unroll
      for (int seg = 0; seg < 2; ++seg) {
        float4 f0 = *(const float4*)(Af + seg * 8);
        float4 f1 = *(const float4*)(Af + seg * 8 + 4);
        v8h hv;
        hv[0] = (half_t)f0.x; hv[1] = (half_t)f0.y; hv[2] = (half_t)f0.z; hv[3] = (half_t)f0.w;
        hv[4] = (half_t)f1.x; hv[5] = (half_t)f1.y; hv[6] = (half_t)f1.z; hv[7] = (half_t)f1.w;
        *(v8h*)&As[arow][akq + seg * 8] = hv;
      }
    }
    // ---- stage B tile: W[n][k] is already n-major over k ----
    {
      const float* Wf = W + (size_t)(n0 + brow) * K + k0 + bkq;
      float4 f0 = *(const float4*)(Wf);
      float4 f1 = *(const float4*)(Wf + 4);
      v8h hv;
      hv[0] = (half_t)f0.x; hv[1] = (half_t)f0.y; hv[2] = (half_t)f0.z; hv[3] = (half_t)f0.w;
      hv[4] = (half_t)f1.x; hv[5] = (half_t)f1.y; hv[6] = (half_t)f1.z; hv[7] = (half_t)f1.w;
      *(v8h*)&Bs[brow][bkq] = hv;
    }
    __syncthreads();

    v16h a0 = load_a_frag(&As[wm * 32][0], 32);
    v16h a1 = load_a_frag(&As[wm * 32 + 16][0], 32);
    v16h b0 = load_b_frag(&Bs[wn * 32][0], 32);
    v16h b1 = load_b_frag(&Bs[wn * 32 + 16][0], 32);
    c00 = wmma_f16(a0, b0, c00);
    c01 = wmma_f16(a0, b1, c01);
    c10 = wmma_f16(a1, b0, c10);
    c11 = wmma_f16(a1, b1, c11);
    __syncthreads();
  }

  // ---- epilogue: C layout -> lane&15 = col, VGPR r -> row r + 8*(lane>>4) ----
  #pragma unroll
  for (int r = 0; r < 8; ++r) {
    const int row0 = m0 + wm * 32 + r + ((lane >> 4) << 3);
    const int row1 = row0 + 16;
    const int col0 = n0 + wn * 32 + (lane & 15);
    const int col1 = col0 + 16;
    const float bv0 = bias[col0], bv1 = bias[col1];
    const float v00 = (c00[r] + bv0) * scale;
    const float v01 = (c01[r] + bv1) * scale;
    const float v10 = (c10[r] + bv0) * scale;
    const float v11 = (c11[r] + bv1) * scale;
    if constexpr (OUT_HALF) {
      half_t* Oh = (half_t*)Outp;
      Oh[(size_t)row0 * N + col0] = (half_t)v00;
      Oh[(size_t)row0 * N + col1] = (half_t)v01;
      Oh[(size_t)row1 * N + col0] = (half_t)v10;
      Oh[(size_t)row1 * N + col1] = (half_t)v11;
    } else {
      float* Of = (float*)Outp;
      Of[(size_t)row0 * N + col0] = v00;
      Of[(size_t)row0 * N + col1] = v01;
      Of[(size_t)row1 * N + col0] = v10;
      Of[(size_t)row1 * N + col1] = v11;
    }
  }
}

// ---------------------------------------------------------------------------
// Fused flash-style attention. Grid: (S/128, H, B); block 256 thr = 8 waves.
// Each wave owns 16 q-rows: online softmax over 1024 keys in chunks of 32.
// K-chunk staged to LDS via the Tensor Data Mover; V staged transposed by VALU.
// ---------------------------------------------------------------------------
__global__ __launch_bounds__(256)
void attn_fused(const half_t* __restrict__ Qh, const half_t* __restrict__ Kh,
                const half_t* __restrict__ Vh, const float* __restrict__ bias,
                const float* __restrict__ amask, const unsigned char* __restrict__ kpm,
                half_t* __restrict__ Ctx) {
  __shared__ __align__(16) half_t Ks[32][64];      // Ks[key][d]
  __shared__ __align__(16) half_t Vt[64][32];      // Vt[d][key] (transposed)
  __shared__ __align__(16) half_t Ps[8][16][32];   // per-wave P tile (C->A relayout)

  const int b = blockIdx.z, h = blockIdx.y;
  const int t = threadIdx.x, lane = t & 31, wave = t >> 5;
  const int qblock = blockIdx.x * 128;
  const int q0 = qblock + wave * 16;

  const int skey = t >> 3;        // 0..31 staging key
  const int sd   = (t & 7) << 3;  // 0..56 staging d-segment

  // Q fragments for this wave's 16 rows, D=64 -> two 16x32 A-frags in regs.
  const half_t* Qbase = Qh + ((size_t)(b * S_) + q0) * E_ + h * D_;
  const v16h aq0 = load_a_frag(Qbase, E_);
  const v16h aq1 = load_a_frag(Qbase + 32, E_);

  float mrow[8], lrow[8];
  #pragma unroll
  for (int r = 0; r < 8; ++r) { mrow[r] = -1e30f; lrow[r] = 0.0f; }
  v8f o0 = {}, o1 = {}, o2 = {}, o3 = {};

  const int qrow_lane = (lane >> 4) << 3;             // +0 or +8 row offset
  const size_t biasBase = ((size_t)b * H_ + h) * S_;  // row base into [B,H,S,S]
#if HAVE_TDM
  const unsigned int ksOff = (unsigned int)(size_t)&Ks[0][0];
#endif

  for (int kc = 0; kc < S_; kc += 32) {
    // ---- stage 32-key chunk: K via TDM DMA, V transposed via VALU ----
#if HAVE_TDM
    if (wave == 0) {
      tdm_load_2d_f16(ksOff, Kh + ((size_t)(b * S_) + kc) * E_ + h * D_,
                      /*tile_d0=*/64, /*tile_d1=*/32, /*stride_d0=*/E_);
      __builtin_amdgcn_s_wait_tensorcnt(0);
    }
#endif
    {
      const size_t gi = ((size_t)(b * S_) + kc + skey) * E_ + h * D_ + sd;
#if !HAVE_TDM
      *(v8h*)&Ks[skey][sd] = *(const v8h*)(Kh + gi);
#endif
      v8h vv = *(const v8h*)(Vh + gi);
      #pragma unroll
      for (int j = 0; j < 8; ++j) Vt[sd + j][skey] = vv[j];
      if (kc + 32 < S_) {  // prefetch next chunk of the streamed bias tensor
        __builtin_prefetch(&bias[(biasBase + qblock + (t & 127)) * S_ + kc + 32], 0, 0);
      }
    }
    __syncthreads();

    // ---- scores S = Q K^T : two 16-key subtiles, two d-steps each ----
    v8f s0 = {}, s1 = {};
    s0 = wmma_f16(aq0, load_b_frag(&Ks[0][0], 64), s0);
    s0 = wmma_f16(aq1, load_b_frag(&Ks[0][0] + 32, 64), s0);
    s1 = wmma_f16(aq0, load_b_frag(&Ks[16][0], 64), s1);
    s1 = wmma_f16(aq1, load_b_frag(&Ks[16][0] + 32, 64), s1);

    // ---- + attn_bias + attn_mask, then key_padding_mask -> NEG_INF ----
    const int key0 = kc + (lane & 15);
    const int key1 = key0 + 16;
    const bool pad0 = kpm[(size_t)b * S_ + key0] != 0;
    const bool pad1 = kpm[(size_t)b * S_ + key1] != 0;
    #pragma unroll
    for (int r = 0; r < 8; ++r) {
      const int qm = r + qrow_lane;
      const size_t brow = (biasBase + q0 + qm) * S_;
      const float bv0 = __builtin_nontemporal_load(&bias[brow + key0]);
      const float bv1 = __builtin_nontemporal_load(&bias[brow + key1]);
      const float t0 = s0[r] + bv0 + amask[(size_t)(q0 + qm) * S_ + key0];
      const float t1 = s1[r] + bv1 + amask[(size_t)(q0 + qm) * S_ + key1];
      s0[r] = pad0 ? NEG_INF_ : t0;
      s1[r] = pad1 ? NEG_INF_ : t1;
    }

    // ---- online softmax (16-lane row reductions; wave32) ----
    #pragma unroll
    for (int r = 0; r < 8; ++r) {
      float v = fmaxf(s0[r], s1[r]);
      v = fmaxf(v, __shfl_xor(v, 1, 32));
      v = fmaxf(v, __shfl_xor(v, 2, 32));
      v = fmaxf(v, __shfl_xor(v, 4, 32));
      v = fmaxf(v, __shfl_xor(v, 8, 32));
      const float mnew  = fmaxf(mrow[r], v);
      const float alpha = __expf(mrow[r] - mnew);
      const float p0 = __expf(s0[r] - mnew);
      const float p1 = __expf(s1[r] - mnew);
      float rs = p0 + p1;
      rs += __shfl_xor(rs, 1, 32);
      rs += __shfl_xor(rs, 2, 32);
      rs += __shfl_xor(rs, 4, 32);
      rs += __shfl_xor(rs, 8, 32);
      lrow[r] = lrow[r] * alpha + rs;
      mrow[r] = mnew;
      o0[r] *= alpha; o1[r] *= alpha; o2[r] *= alpha; o3[r] *= alpha;
      const int qm = r + qrow_lane;
      Ps[wave][qm][lane & 15]        = (half_t)p0;   // C-layout -> LDS
      Ps[wave][qm][16 + (lane & 15)] = (half_t)p1;
    }

    // ---- O += P V : reload P in A-layout (same-wave LDS, no barrier) ----
    const v16h pa = load_a_frag(&Ps[wave][0][0], 32);
    o0 = wmma_f16(pa, load_b_frag(&Vt[0][0],  32), o0);
    o1 = wmma_f16(pa, load_b_frag(&Vt[16][0], 32), o1);
    o2 = wmma_f16(pa, load_b_frag(&Vt[32][0], 32), o2);
    o3 = wmma_f16(pa, load_b_frag(&Vt[48][0], 32), o3);
    __syncthreads();
  }

  // ---- epilogue: O / l -> ctx f16 [B,S,H,D] ----
  #pragma unroll
  for (int r = 0; r < 8; ++r) {
    const int qm = r + qrow_lane;
    const float inv = 1.0f / lrow[r];
    const size_t orow = ((size_t)(b * S_) + q0 + qm) * E_ + h * D_;
    Ctx[orow + (lane & 15)]      = (half_t)(o0[r] * inv);
    Ctx[orow + 16 + (lane & 15)] = (half_t)(o1[r] * inv);
    Ctx[orow + 32 + (lane & 15)] = (half_t)(o2[r] * inv);
    Ctx[orow + 48 + (lane & 15)] = (half_t)(o3[r] * inv);
  }
}

extern "C" void kernel_launch(void* const* d_in, const int* in_sizes, int n_in,
                              void* d_out, int out_size, void* d_ws, size_t ws_size,
                              hipStream_t stream) {
  (void)in_sizes; (void)n_in; (void)out_size; (void)ws_size;
  const float* query       = (const float*)d_in[0];
  const float* attn_bias   = (const float*)d_in[3];
  const unsigned char* kpm = (const unsigned char*)d_in[4];
  const float* amask       = (const float*)d_in[5];
  const float* Wq = (const float*)d_in[9];
  const float* bq = (const float*)d_in[10];
  const float* Wk = (const float*)d_in[11];
  const float* bk = (const float*)d_in[12];
  const float* Wv = (const float*)d_in[13];
  const float* bv = (const float*)d_in[14];
  const float* Wo = (const float*)d_in[15];
  const float* bo = (const float*)d_in[16];

  const int M = B_ * S_, N = E_, K = E_;

  half_t* Qh  = (half_t*)d_ws;                  // 16 MB each
  half_t* Kh  = Qh + (size_t)M * E_;
  half_t* Vh  = Kh + (size_t)M * E_;
  half_t* Ctx = Vh + (size_t)M * E_;

  dim3 gg(M / 128, N / 64), bb(256);
  const float scl = 0.125f;  // D^-0.5, applied after bias (matches reference)
  gemm_xwt<false, true><<<gg, bb, 0, stream>>>(query, Wq, bq, scl,  Qh, M, N, K);
  gemm_xwt<false, true><<<gg, bb, 0, stream>>>(query, Wk, bk, 1.0f, Kh, M, N, K);
  gemm_xwt<false, true><<<gg, bb, 0, stream>>>(query, Wv, bv, 1.0f, Vh, M, N, K);

  dim3 ga(S_ / 128, H_, B_);
  attn_fused<<<ga, bb, 0, stream>>>(Qh, Kh, Vh, attn_bias, amask, kpm, Ctx);

  gemm_xwt<true, false><<<gg, bb, 0, stream>>>(Ctx, Wo, bo, 1.0f, d_out, M, N, K);
}
